// QuerySAT_87797721465347
// MI455X (gfx1250) — matrix operand order repaired and checked by reading, not compiled, not run on hardware
//
#include <hip/hip_runtime.h>
#include <hip/hip_bf16.h>
#include <math.h>

typedef __attribute__((ext_vector_type(16))) _Float16 v16h;
typedef __attribute__((ext_vector_type(8)))  _Float16 v8h;
typedef __attribute__((ext_vector_type(8)))  float    v8f;

#define NV 50000
#define NC 200000
#define NL 600000
#define NG 100
#define FM 128
#define QM 32

// ---------------------------------------------------------------------------
// Tiled WMMA GEMM: C[M,N] = act(A[M,Kp] (f16) @ Wt (f16, [N][Kp]) + bias)
// Block tile: 256 (M) x 32 (N), K step 32, ping-pong LDS (1 barrier / K-step).
// 8 waves as 4 (M) x 2 (N); each wave owns a 64x16 patch = 4 f32 accumulators.
// An inline-asm fence that (re)defines all fragment registers pins every
// ds_load before the 4 back-to-back v_wmma_f32_16x16x32_f16 issues, so a
// single ds-wait covers the whole WMMA burst.
// Fragment addressing follows the CDNA5 wave32 VGPR layouts (ISA 7.12.2).
// ---------------------------------------------------------------------------
__global__ __launch_bounds__(256) void k_gemm(
    const _Float16* __restrict__ A, int lda,
    const _Float16* __restrict__ Wt,           // [N][Kp], k contiguous
    const float*    __restrict__ bias,
    float*          __restrict__ C32, int ldc32,
    _Float16*       __restrict__ C16, int ldc16,
    int M, int N, int Kp, int relu)
{
  __shared__ _Float16 sA[2][256 * 32];
  __shared__ _Float16 sB[2][32 * 32];

  const int tid  = threadIdx.x;
  const int lane = tid & 31;
  const int wave = tid >> 5;
  const int wm   = wave & 3;    // 0..3 -> 64-row sub-tile
  const int wn   = wave >> 2;   // 0..1 -> 16-col sub-tile
  const int m0   = blockIdx.x * 256;
  const int n0   = blockIdx.y * 32;
  const int half = lane >> 4;   // K-half selector
  const int mr   = lane & 15;   // row (A) / col (B) within fragment
  const bool interior = (m0 + 256 <= M);

  v8f acc[4] = {{}, {}, {}, {}};
  union U16 { v16h v; v8h h[2]; };
  v8h ra[4]; v8h rb = {};

  auto load_tile = [&](int k0) {
    if (interior) {
      #pragma unroll
      for (int i = 0; i < 4; ++i) {
        int c = tid + i * 256;
        int r = c >> 2, c8 = (c & 3) * 8;
        ra[i] = *(const v8h*)(A + (size_t)(m0 + r) * lda + k0 + c8);
      }
    } else {
      #pragma unroll
      for (int i = 0; i < 4; ++i) {
        int c = tid + i * 256;
        int r = c >> 2, c8 = (c & 3) * 8;
        v8h v = {};
        if (m0 + r < M) v = *(const v8h*)(A + (size_t)(m0 + r) * lda + k0 + c8);
        ra[i] = v;
      }
    }
    if (tid < 128) {
      int n = tid >> 2, c8 = (tid & 3) * 8;
      rb = *(const v8h*)(Wt + (size_t)(n0 + n) * Kp + k0 + c8);
    }
  };
  auto store_tile = [&](int pp) {
    #pragma unroll
    for (int i = 0; i < 4; ++i) {
      int c = tid + i * 256;
      int r = c >> 2, c8 = (c & 3) * 8;
      *(v8h*)(sA[pp] + r * 32 + c8) = ra[i];
    }
    if (tid < 128) {
      int n = tid >> 2, c8 = (tid & 3) * 8;
      *(v8h*)(sB[pp] + n * 32 + c8) = rb;
    }
  };

  int p = 0;
  load_tile(0);
  store_tile(0);
  __syncthreads();

  for (int k0 = 0; k0 < Kp; k0 += 32) {
    const bool last = (k0 + 32 >= Kp);
    if (!last) {
      load_tile(k0 + 32);                   // global -> regs (next tile)
      if (interior && (k0 + 64 < Kp))       // single uniform-guarded L2 warm
        __builtin_prefetch(
            A + (size_t)(m0 + (tid >> 2)) * lda + k0 + 64 + (tid & 3) * 8, 0, 1);
    }
    // ---- load ALL fragments to VGPRs first ----
    // B fragment 32x16: lane=col; VGPR0-3 K=half*8.., VGPR4-7 K=16+half*8..
    U16 b, a0, a1, a2, a3;
    { const _Float16* bp = sB[p] + (wn * 16 + mr) * 32 + half * 8;
      b.h[0] = *(const v8h*)(bp);
      b.h[1] = *(const v8h*)(bp + 16); }
    // A fragments 16x32: lane=row; VGPR0-3 K=half*8.., VGPR4-7 K=16+half*8..
    const _Float16* abase = sA[p] + (wm * 64 + mr) * 32 + half * 8;
    a0.h[0] = *(const v8h*)(abase);
    a0.h[1] = *(const v8h*)(abase + 16);
    a1.h[0] = *(const v8h*)(abase + 512);
    a1.h[1] = *(const v8h*)(abase + 512 + 16);
    a2.h[0] = *(const v8h*)(abase + 1024);
    a2.h[1] = *(const v8h*)(abase + 1024 + 16);
    a3.h[0] = *(const v8h*)(abase + 1536);
    a3.h[1] = *(const v8h*)(abase + 1536 + 16);
    // Fence that (re)defines every fragment: all ds_loads must complete before
    // it, and no WMMA (consumer of these values) can be hoisted above it.
    asm volatile("" : "+v"(a0.v), "+v"(a1.v), "+v"(a2.v), "+v"(a3.v), "+v"(b.v));
    // ---- 4 back-to-back WMMAs ----
    acc[0] = __builtin_amdgcn_wmma_f32_16x16x32_f16(false, a0.v, false, b.v,
                                                    (short)0, acc[0], false, false);
    acc[1] = __builtin_amdgcn_wmma_f32_16x16x32_f16(false, a1.v, false, b.v,
                                                    (short)0, acc[1], false, false);
    acc[2] = __builtin_amdgcn_wmma_f32_16x16x32_f16(false, a2.v, false, b.v,
                                                    (short)0, acc[2], false, false);
    acc[3] = __builtin_amdgcn_wmma_f32_16x16x32_f16(false, a3.v, false, b.v,
                                                    (short)0, acc[3], false, false);
    if (!last) store_tile(p ^ 1);           // regs -> other LDS buffer
    __syncthreads();
    p ^= 1;
  }

  const int col = n0 + wn * 16 + mr;
  const float bs = bias ? bias[col] : 0.f;
  #pragma unroll
  for (int f = 0; f < 4; ++f) {
    #pragma unroll
    for (int j = 0; j < 8; ++j) {
      int r = m0 + wm * 64 + f * 16 + half * 8 + j;
      float v = acc[f][j] + bs;
      if (relu) v = fmaxf(v, 0.f);
      if (r < M) {
        if (C32) C32[(size_t)r * ldc32 + col] = v;
        if (C16) C16[(size_t)r * ldc16 + col] = (_Float16)v;
      }
    }
  }
}

// ---- weight conversion: W[K][N] f32 -> Wt[Np][Kp] f16 (transposed, padded) ----
__global__ void k_wconv(const float* __restrict__ W, _Float16* __restrict__ Wt,
                        int K, int N, int Kp, int Np) {
  int i = blockIdx.x * 256 + threadIdx.x;
  if (i >= Np * Kp) return;
  int n = i / Kp, k = i % Kp;
  float v = (k < K && n < N) ? W[(size_t)k * N + n] : 0.f;
  Wt[i] = (_Float16)v;
}

__global__ void k_padbias(const float* b, float* bp, int n, int np) {
  int i = threadIdx.x;
  if (i < np) bp[i] = (i < n) ? b[i] : 0.f;
}

__global__ void k_zero(float* p, size_t n) {
  size_t i = (size_t)blockIdx.x * 256 + threadIdx.x;
  if (i < n) p[i] = 0.f;
}

// zero_state: (one_hot(0) - 1/F) * sqrt(F)*0.25
__global__ void k_init_state(float* s, size_t n) {
  size_t i = (size_t)blockIdx.x * 256 + threadIdx.x;
  if (i >= n) return;
  int c = (int)(i & (FM - 1));
  const float base = 11.3137085f * 0.25f;  // sqrt(128)*0.25
  s[i] = (c == 0 ? (1.f - 1.f / FM) : (-1.f / FM)) * base;
}

// ---- concat builders (write full padded row incl. zero pad, f16) ----
__global__ void k_build_v1(const float* __restrict__ vars,
                           const float* __restrict__ nz,
                           _Float16* __restrict__ A) {
  size_t i = (size_t)blockIdx.x * 256 + threadIdx.x;
  if (i >= (size_t)NV * 160) return;
  int r = (int)(i / 160), c = (int)(i % 160);
  float v = (c < 128) ? vars[(size_t)r * FM + c]
          : (c < 132) ? nz[(size_t)r * 4 + (c - 128)] : 0.f;
  A[i] = (_Float16)v;
}

__global__ void k_build_clause_unit(const float* __restrict__ cst,
                                    const float* __restrict__ cl,
                                    _Float16* __restrict__ A) {
  size_t i = (size_t)blockIdx.x * 256 + threadIdx.x;
  if (i >= (size_t)NC * 160) return;
  int r = (int)(i / 160), c = (int)(i % 160);
  float v = (c < 128) ? cst[(size_t)r * FM + c] : cl[(size_t)r * QM + (c - 128)];
  A[i] = (_Float16)v;
}

__global__ void k_build_unit(const float* __restrict__ vars,
                             const float* __restrict__ vg,
                             const float* __restrict__ lp,
                             const float* __restrict__ ln,
                             _Float16* __restrict__ A) {
  size_t i = (size_t)blockIdx.x * 256 + threadIdx.x;
  if (i >= (size_t)NV * 224) return;
  int r = (int)(i / 224), c = (int)(i % 224);
  float v;
  if      (c < 128) v = vars[(size_t)r * FM + c];
  else if (c < 160) v = vg[(size_t)r * QM + (c - 128)];
  else if (c < 192) v = lp[(size_t)r * QM + (c - 160)];
  else              v = ln[(size_t)r * QM + (c - 192)];
  A[i] = (_Float16)v;
}

__global__ void k_f32tof16(const float* __restrict__ s, _Float16* __restrict__ d,
                           size_t n) {
  size_t i = (size_t)blockIdx.x * 256 + threadIdx.x;
  if (i < n) d[i] = (_Float16)s[i];
}

// ---- literal kernels: one lane per (literal, query map) ----
__device__ __forceinline__ float softplusf(float x) {
  return x > 20.f ? x : log1pf(__expf(x));
}

__global__ __launch_bounds__(256) void k_lit_sp(
    const int* __restrict__ lits, const int* __restrict__ lc,
    const float* __restrict__ q, float* __restrict__ clS) {
  int w = (blockIdx.x * 256 + threadIdx.x) >> 5;
  int j = threadIdx.x & 31;
  if (w >= NL) return;
  int L = lits[w];
  int v = (L < 0 ? -L : L) - 1;
  float s = L < 0 ? -1.f : 1.f;
  float val = q[(size_t)v * QM + j] * s;
  atomicAdd(clS + (size_t)lc[w] * QM + j, softplusf(val));
}

__global__ void k_exp_neg(float* p, size_t n) {
  size_t i = (size_t)blockIdx.x * 256 + threadIdx.x;
  if (i < n) p[i] = __expf(-p[i]);
}

__global__ __launch_bounds__(256) void k_lit_grad(
    const int* __restrict__ lits, const int* __restrict__ lc,
    const float* __restrict__ q, const float* __restrict__ cl,
    float* __restrict__ vg) {
  int w = (blockIdx.x * 256 + threadIdx.x) >> 5;
  int j = threadIdx.x & 31;
  if (w >= NL) return;
  int L = lits[w];
  int v = (L < 0 ? -L : L) - 1;
  float s = L < 0 ? -1.f : 1.f;
  float val = q[(size_t)v * QM + j] * s;
  float sig = 1.f / (1.f + __expf(-val));
  float g = -cl[(size_t)lc[w] * QM + j] * sig * s;
  atomicAdd(vg + (size_t)v * QM + j, g);
}

__global__ __launch_bounds__(256) void k_lit_gather(
    const int* __restrict__ lits, const int* __restrict__ lc,
    const float* __restrict__ cdata,   // [NC,160], cols 0..31
    float* __restrict__ lp, float* __restrict__ ln) {
  int w = (blockIdx.x * 256 + threadIdx.x) >> 5;
  int j = threadIdx.x & 31;
  if (w >= NL) return;
  int L = lits[w];
  int v = (L < 0 ? -L : L) - 1;
  float g = cdata[(size_t)lc[w] * 160 + j];
  atomicAdd((L > 0 ? lp : ln) + (size_t)v * QM + j, g);
}

// ---- pair-norm (graphs are contiguous row blocks): 1 block/graph, col/thread ----
__global__ __launch_bounds__(128) void k_pairnorm(
    const float* __restrict__ src, int lds, int colOff,
    float* __restrict__ state, int R) {
  int g = blockIdx.x, c = threadIdx.x;
  const float* s = src + (size_t)g * R * lds + colOff + c;
  __shared__ float sred[128];
  float sum = 0.f;
  for (int r = 0; r < R; ++r) sum += s[(size_t)r * lds];
  float mean = sum / (float)R;
  float vs = 0.f;
  for (int r = 0; r < R; ++r) { float d = s[(size_t)r * lds] - mean; vs += d * d; }
  sred[c] = vs;
  __syncthreads();
  for (int o = 64; o > 0; o >>= 1) { if (c < o) sred[c] += sred[c + o]; __syncthreads(); }
  float var = sred[0] / ((float)R * 128.f);
  float scale = rsqrtf(var + 1e-6f) * 0.25f;
  float* st = state + (size_t)g * R * FM + c;
  for (int r = 0; r < R; ++r) {
    float d = (s[(size_t)r * lds] - mean) * scale;
    st[(size_t)r * FM] = d + 0.1f * st[(size_t)r * FM];
  }
}

// ---- per-round loss ----
__global__ void k_lit_final(const int* __restrict__ lits, const int* __restrict__ lc,
                            const float* __restrict__ logits, float* __restrict__ spS) {
  int i = blockIdx.x * 256 + threadIdx.x;
  if (i >= NL) return;
  int L = lits[i];
  int v = (L < 0 ? -L : L) - 1;
  float s = L < 0 ? -1.f : 1.f;
  atomicAdd(spS + lc[i], softplusf(logits[(size_t)v * 32] * s));
}

__global__ void k_clause_final(const float* __restrict__ spS,
                               const int* __restrict__ cmask,
                               float* __restrict__ perGraph) {
  int i = blockIdx.x * 256 + threadIdx.x;
  if (i >= NC) return;
  float cv = __expf(-spS[i]);
  float pc = cv * (-__logf(1.f - cv + 1e-10f));
  atomicAdd(perGraph + cmask[i], pc);
}

__global__ __launch_bounds__(128) void k_graph_loss(const float* __restrict__ perGraph,
                                                    float* __restrict__ acc) {
  __shared__ float sred[128];
  int t = threadIdx.x;
  sred[t] = (t < NG) ? sqrtf(perGraph[t] + 1e-6f) : 0.f;
  __syncthreads();
  for (int o = 64; o > 0; o >>= 1) { if (t < o) sred[t] += sred[t + o]; __syncthreads(); }
  if (t == 0) atomicAdd(acc, sred[0]);
}

__global__ void k_finalize(const float* __restrict__ logits,
                           const float* __restrict__ acc, float* __restrict__ out) {
  int i = blockIdx.x * 256 + threadIdx.x;
  if (i < NV)       out[i] = logits[(size_t)i * 32];
  else if (i == NV) out[i] = acc[0] * 0.125f;
}

// ---------------------------------------------------------------------------
extern "C" void kernel_launch(void* const* d_in, const int* in_sizes, int n_in,
                              void* d_out, int out_size, void* d_ws, size_t ws_size,
                              hipStream_t stream) {
  // param order: q(W0,b0,W1,b1,W2,b2), c(...), u(...), o(...), noise, lits,
  //              lit_clause, variables_mask, clauses_mask
  const float* pW[12]; const float* pB[12];
  for (int l = 0; l < 12; ++l) { pW[l] = (const float*)d_in[2*l]; pB[l] = (const float*)d_in[2*l+1]; }
  const float* noise = (const float*)d_in[24];
  const int* lits  = (const int*)d_in[25];
  const int* litcl = (const int*)d_in[26];
  const int* cmask = (const int*)d_in[28];
  float* out = (float*)d_out;

  const int LK[12]  = {132, 64, 64, 160, 384, 384, 224, 256, 256, 128, 128, 128};
  const int LN[12]  = { 64, 64, 32, 384, 384, 160, 256, 256, 128, 128, 128,   1};
  const int LKP[12] = {160, 64, 64, 160, 384, 384, 224, 256, 256, 128, 128, 128};
  const int LNP[12] = { 64, 64, 32, 384, 384, 160, 256, 256, 128, 128, 128,  32};

  size_t off = 0;
  char* ws = (char*)d_ws;
  auto alloc = [&](size_t bytes) -> char* {
    char* p = ws + off; off = (off + bytes + 255) & ~(size_t)255; return p;
  };
  _Float16* wt[12];
  for (int l = 0; l < 12; ++l)
    wt[l] = (_Float16*)alloc((size_t)LKP[l] * LNP[l] * sizeof(_Float16));
  float* bo2   = (float*)alloc(32 * sizeof(float));
  float* vars  = (float*)alloc((size_t)NV * FM * sizeof(float));
  float* cst   = (float*)alloc((size_t)NC * FM * sizeof(float));
  _Float16* bufA = (_Float16*)alloc((size_t)NC * 384 * sizeof(_Float16));
  _Float16* bufB = (_Float16*)alloc((size_t)NC * 384 * sizeof(_Float16));
  float* query = (float*)alloc((size_t)NV * QM * sizeof(float));
  float* clLs  = (float*)alloc((size_t)NC * QM * sizeof(float));
  float* vgrad = (float*)alloc((size_t)NV * QM * sizeof(float));
  float* lpos  = (float*)alloc((size_t)NV * QM * sizeof(float));
  float* lneg  = (float*)alloc((size_t)NV * QM * sizeof(float));
  float* cdata = (float*)alloc((size_t)NC * 160 * sizeof(float));
  float* gate  = (float*)alloc((size_t)NV * FM * sizeof(float));
  float* logit = (float*)alloc((size_t)NV * 32 * sizeof(float));
  float* spS   = (float*)alloc((size_t)NC * sizeof(float));
  float* perG  = (float*)alloc(NG * sizeof(float));
  float* lacc  = (float*)alloc(sizeof(float));

  auto blocks = [](size_t n) { return (unsigned)((n + 255) / 256); };
  auto gemm = [&](const _Float16* A, int lda, int layer,
                  const float* bias, float* c32, int ldc32,
                  _Float16* c16, int ldc16, int M, int relu) {
    dim3 g((M + 255) / 256, LNP[layer] / 32);
    k_gemm<<<g, 256, 0, stream>>>(A, lda, wt[layer], bias, c32, ldc32,
                                  c16, ldc16, M, LNP[layer], LKP[layer], relu);
  };

  for (int l = 0; l < 12; ++l) {
    int n = LKP[l] * LNP[l];
    k_wconv<<<blocks(n), 256, 0, stream>>>(pW[l], wt[l], LK[l], LN[l], LKP[l], LNP[l]);
  }
  k_padbias<<<1, 32, 0, stream>>>(pB[11], bo2, 1, 32);
  k_init_state<<<blocks((size_t)NV * FM), 256, 0, stream>>>(vars, (size_t)NV * FM);
  k_init_state<<<blocks((size_t)NC * FM), 256, 0, stream>>>(cst, (size_t)NC * FM);
  k_zero<<<1, 256, 0, stream>>>(lacc, 1);

  for (int r = 0; r < 8; ++r) {
    const float* nz = noise + (size_t)r * NV * 4;
    // --- variables_query MLP ---
    k_build_v1<<<blocks((size_t)NV * 160), 256, 0, stream>>>(vars, nz, bufA);
    gemm(bufA, 160, 0, pB[0], nullptr, 0, bufB, 64, NV, 1);
    gemm(bufB,  64, 1, pB[1], nullptr, 0, bufA, 64, NV, 1);
    gemm(bufA,  64, 2, pB[2], query, QM, nullptr, 0, NV, 0);
    // --- softplus clause loss + analytic grad (value_and_grad) ---
    k_zero<<<blocks((size_t)NC * QM), 256, 0, stream>>>(clLs, (size_t)NC * QM);
    k_lit_sp<<<blocks((size_t)NL * 32), 256, 0, stream>>>(lits, litcl, query, clLs);
    k_exp_neg<<<blocks((size_t)NC * QM), 256, 0, stream>>>(clLs, (size_t)NC * QM);
    k_zero<<<blocks((size_t)NV * QM), 256, 0, stream>>>(vgrad, (size_t)NV * QM);
    k_lit_grad<<<blocks((size_t)NL * 32), 256, 0, stream>>>(lits, litcl, query, clLs, vgrad);
    // --- clause MLP + pair-norm state update ---
    k_build_clause_unit<<<blocks((size_t)NC * 160), 256, 0, stream>>>(cst, clLs, bufA);
    gemm(bufA, 160, 3, pB[3], nullptr, 0, bufB, 384, NC, 1);
    gemm(bufB, 384, 4, pB[4], nullptr, 0, bufA, 384, NC, 1);
    gemm(bufA, 384, 5, pB[5], cdata, 160, nullptr, 0, NC, 0);
    k_pairnorm<<<NG, 128, 0, stream>>>(cdata, 160, QM, cst, NC / NG);
    // --- gather clause messages to variables ---
    k_zero<<<blocks((size_t)NV * QM), 256, 0, stream>>>(lpos, (size_t)NV * QM);
    k_zero<<<blocks((size_t)NV * QM), 256, 0, stream>>>(lneg, (size_t)NV * QM);
    k_lit_gather<<<blocks((size_t)NL * 32), 256, 0, stream>>>(lits, litcl, cdata, lpos, lneg);
    // --- update_gate MLP + pair-norm variable update ---
    k_build_unit<<<blocks((size_t)NV * 224), 256, 0, stream>>>(vars, vgrad, lpos, lneg, bufA);
    gemm(bufA, 224, 6, pB[6], nullptr, 0, bufB, 256, NV, 1);
    gemm(bufB, 256, 7, pB[7], nullptr, 0, bufA, 256, NV, 1);
    gemm(bufA, 256, 8, pB[8], gate, FM, nullptr, 0, NV, 0);
    k_pairnorm<<<NG, 128, 0, stream>>>(gate, FM, 0, vars, NV / NG);
    // --- output MLP + per-round loss ---
    k_f32tof16<<<blocks((size_t)NV * FM), 256, 0, stream>>>(vars, bufA, (size_t)NV * FM);
    gemm(bufA, 128,  9, pB[9],  nullptr, 0, bufB, 128, NV, 1);
    gemm(bufB, 128, 10, pB[10], nullptr, 0, bufA, 128, NV, 1);
    gemm(bufA, 128, 11, bo2, logit, 32, nullptr, 0, NV, 0);
    k_zero<<<blocks(NC), 256, 0, stream>>>(spS, NC);
    k_lit_final<<<blocks(NL), 256, 0, stream>>>(lits, litcl, logit, spS);
    k_zero<<<1, 256, 0, stream>>>(perG, NG);
    k_clause_final<<<blocks(NC), 256, 0, stream>>>(spS, cmask, perG);
    k_graph_loss<<<1, 128, 0, stream>>>(perG, lacc);
  }
  k_finalize<<<blocks(NV + 1), 256, 0, stream>>>(logit, lacc, out);
  (void)in_sizes; (void)n_in; (void)out_size; (void)ws_size;
}
